// WaveAttention_8383776162028
// MI455X (gfx1250) — matrix operand order repaired and checked by reading, not compiled, not run on hardware
//
#include <hip/hip_runtime.h>
#include <hip/hip_bf16.h>

// ---------------------------------------------------------------------------
// WaveAttention pipeline for gfx1250 (MI455X).  All dense contractions run on
// v_wmma_f32_16x16x32_f16 (wave32 WMMA).  Attention is fused (196 keys ->
// padded 224, scores live in LDS) so the 157MB logits tensor never exists.
// ---------------------------------------------------------------------------

typedef __attribute__((ext_vector_type(16))) _Float16 v16h;
typedef __attribute__((ext_vector_type(8)))  float    v8f;

#define BATCH   8
#define HH      56
#define WW      56
#define CC      512
#define NN      3136            // 56*56
#define C4      128
#define NHEAD   8
#define HD      64
#define H2      28
#define NKEY    196             // 14*14
#define KPAD    224             // 196 padded to 7*32
#define EPS_BN  1e-5f
#define EPS_LN  1e-5f

__device__ __forceinline__ v8f zero8() {
  v8f z = {0.f, 0.f, 0.f, 0.f, 0.f, 0.f, 0.f, 0.f};
  return z;
}

__device__ __forceinline__ v8f wmma_f16(v16h a, v16h b, v8f c) {
  // D(16x16,f32) = A(16x32,f16) * B(32x16,f16) + C
  return __builtin_amdgcn_wmma_f32_16x16x32_f16(
      /*neg_a=*/false, a, /*neg_b=*/false, b,
      /*c_mod=*/(short)0, c, /*reuse_a=*/false, /*reuse_b=*/false);
}

// ---- A fragment: 16x32 f16, row-major, row = lane&15.
// Layout: lanes 0-15 hold K 0..15, lanes 16-31 K 16..31 (contiguous, matches B).
__device__ __forceinline__ v16h fragA_lanerow(const _Float16* lanerow, bool valid) {
  const int lane = threadIdx.x & 31;
  const _Float16* q = lanerow + ((lane >> 4) << 4);
  v16h f;
#pragma unroll
  for (int e = 0; e < 8; ++e) f[e]     = valid ? q[e]      : (_Float16)0.f;
#pragma unroll
  for (int e = 0; e < 8; ++e) f[8 + e] = valid ? q[8 + e] : (_Float16)0.f;
  return f;
}

__device__ __forceinline__ v16h fragA_ld(const _Float16* p /*at (m0,k0)*/, int ld) {
  const int lane = threadIdx.x & 31;
  return fragA_lanerow(p + (size_t)(lane & 15) * ld, true);
}

// ---- B fragment: 32x16 f16 (KxN); lane holds column n = lane&15,
// elements e -> K = 16*(lane>>4) + e.  Operand supplied as W stored (N x K)
// row-major (weight-transposed GEMM): per-lane contiguous 16 halves.
__device__ __forceinline__ v16h fragB_nk(const _Float16* p /*at (n0,k0)*/, int ld) {
  const int lane = threadIdx.x & 31;
  const _Float16* q = p + (size_t)(lane & 15) * ld + ((lane >> 4) << 4);
  v16h f;
#pragma unroll
  for (int e = 0; e < 16; ++e) f[e] = q[e];
  return f;
}

__device__ __forceinline__ v16h fragB_nk_guard(const _Float16* p, int ld, int n0, int nlim) {
  const int lane = threadIdx.x & 31;
  const int nr = lane & 15;
  const bool v = (n0 + nr) < nlim;
  const _Float16* q = p + (size_t)nr * ld + ((lane >> 4) << 4);
  v16h f;
#pragma unroll
  for (int e = 0; e < 16; ++e) f[e] = v ? q[e] : (_Float16)0.f;
  return f;
}

// B operand stored (K x N) row-major (e.g. V matrix): strided per-element.
__device__ __forceinline__ v16h fragB_kn_guard(const _Float16* p /*at (kbase,n0)*/,
                                               int ld, int kbase, int klim) {
  const int lane = threadIdx.x & 31;
  const int n  = lane & 15;
  const int k0 = (lane >> 4) << 4;
  v16h f;
#pragma unroll
  for (int e = 0; e < 16; ++e) {
    const int k = k0 + e;
    f[e] = (kbase + k < klim) ? p[(size_t)k * ld + n] : (_Float16)0.f;
  }
  return f;
}

// ---------------------------------------------------------------------------
// Generic GEMM: C[M,N] = A[M,K](f16,row-major) * W[N,K](f16,row-major)^T
// mode 0: f32 out, +bias     mode 1: f16 out, +bias
// mode 2: f16 out, relu((acc+bias)*bn_scale + bn_beta)
// block = 128 threads (4 waves); wave w -> n-tile blockIdx.y*4+w.
// ---------------------------------------------------------------------------
__global__ __launch_bounds__(128) void gemm_xwt(
    const _Float16* __restrict__ A, const _Float16* __restrict__ W,
    const float* __restrict__ bias, const float* __restrict__ g,
    const float* __restrict__ be, float* __restrict__ Cf,
    _Float16* __restrict__ Ch, int K, int ldc, int mode) {
  const int lane = threadIdx.x & 31;
  const int wv   = threadIdx.x >> 5;
  const size_t m0 = (size_t)blockIdx.x * 16;
  const int    n0 = (blockIdx.y * 4 + wv) * 16;
  const _Float16* Ap = A + m0 * K;
  const _Float16* Wp = W + (size_t)n0 * K;
  v8f acc = zero8();
  for (int k = 0; k < K; k += 32) {
    v16h a = fragA_ld(Ap + k, K);
    v16h w = fragB_nk(Wp + k, K);
    acc = wmma_f16(a, w, acc);
  }
  const int n = n0 + (lane & 15);
  const float bb = bias[n];
  float s = 1.f, bt = 0.f;
  if (mode == 2) { s = g[n] * rsqrtf(1.f + EPS_BN); bt = be[n]; }
#pragma unroll
  for (int r = 0; r < 8; ++r) {
    const size_t m = m0 + r + ((lane >> 4) << 3);
    float v = acc[r] + bb;
    if (mode == 2) v = fmaxf(v * s + bt, 0.f);
    if (mode == 0) Cf[m * ldc + n] = v;
    else           Ch[m * ldc + n] = (_Float16)v;
  }
}

// ---------------------------------------------------------------------------
// 3x3 conv (pad 1) as implicit-im2col WMMA GEMM over NHWC f16 input
// (B,28,28,512) with repacked weights Wt[tap][O][I]; BN+ReLU epilogue -> f16.
// ---------------------------------------------------------------------------
__global__ __launch_bounds__(128) void conv3x3_bn_relu(
    const _Float16* __restrict__ in, const _Float16* __restrict__ Wt,
    const float* __restrict__ bias, const float* __restrict__ g,
    const float* __restrict__ be, _Float16* __restrict__ out) {
  const int lane = threadIdx.x & 31;
  const int wv   = threadIdx.x >> 5;
  const int b    = blockIdx.z;
  const int p0   = blockIdx.x * 16;
  const int n0   = (blockIdx.y * 4 + wv) * 16;
  const int p    = p0 + (lane & 15);
  const int y = p / H2, x = p % H2;
  v8f acc = zero8();
#pragma unroll 1
  for (int t = 0; t < 9; ++t) {
    const int sy = y + t / 3 - 1;
    const int sx = x + t % 3 - 1;
    const bool valid = ((unsigned)sy < (unsigned)H2) && ((unsigned)sx < (unsigned)H2);
    const int syc = valid ? sy : 0, sxc = valid ? sx : 0;
    const _Float16* lanerow = in + (((size_t)b * H2 + syc) * H2 + sxc) * CC;
    const _Float16* wrow = Wt + ((size_t)t * CC + n0) * CC;
    for (int k = 0; k < CC; k += 32) {
      v16h a = fragA_lanerow(lanerow + k, valid);
      v16h w = fragB_nk(wrow + k, CC);
      acc = wmma_f16(a, w, acc);
    }
  }
  const int n = n0 + (lane & 15);
  const float s  = g[n] * rsqrtf(1.f + EPS_BN);
  const float bb = bias[n], bt = be[n];
#pragma unroll
  for (int r = 0; r < 8; ++r) {
    const int m = p0 + r + ((lane >> 4) << 3);
    const float v = fmaxf((acc[r] + bb) * s + bt, 0.f);
    out[((size_t)b * (H2 * H2) + m) * CC + n] = (_Float16)v;
  }
}

// ---------------------------------------------------------------------------
// 2x2 stride-2 conv (kv embed) as WMMA GEMM; input (B,28,28,512) f16 NHWC,
// Wt[tap][O][I]; +bias -> f32 kv_in (B*196, 512).
// ---------------------------------------------------------------------------
__global__ __launch_bounds__(128) void conv2x2_s2_bias(
    const _Float16* __restrict__ in, const _Float16* __restrict__ Wt,
    const float* __restrict__ bias, float* __restrict__ out) {
  const int lane = threadIdx.x & 31;
  const int wv   = threadIdx.x >> 5;
  const int b    = blockIdx.z;
  const int p0   = blockIdx.x * 16;
  const int n0   = (blockIdx.y * 4 + wv) * 16;
  const int p    = p0 + (lane & 15);
  const bool rowok = p < NKEY;
  const int pc = rowok ? p : 0;
  const int y = pc / 14, x = pc % 14;
  v8f acc = zero8();
#pragma unroll 1
  for (int t = 0; t < 4; ++t) {
    const int sy = 2 * y + (t >> 1);
    const int sx = 2 * x + (t & 1);
    const _Float16* lanerow = in + (((size_t)b * H2 + sy) * H2 + sx) * CC;
    const _Float16* wrow = Wt + ((size_t)t * CC + n0) * CC;
    for (int k = 0; k < CC; k += 32) {
      v16h a = fragA_lanerow(lanerow + k, rowok);
      v16h w = fragB_nk(wrow + k, CC);
      acc = wmma_f16(a, w, acc);
    }
  }
  const int n = n0 + (lane & 15);
  const float bb = bias[n];
#pragma unroll
  for (int r = 0; r < 8; ++r) {
    const int m = p0 + r + ((lane >> 4) << 3);
    if (m < NKEY) out[((size_t)b * NKEY + m) * CC + n] = acc[r] + bb;
  }
}

// ---------------------------------------------------------------------------
// Fused attention: one wave per (b, h, 16-query tile).  196 keys padded to
// 224.  QK^T (wmma) -> LDS f32 -> softmax -> P(f16) -> P*V (wmma) -> f16
// straight into concat buffer columns [0,512).
// ---------------------------------------------------------------------------
__global__ __launch_bounds__(32) void attention_fused(
    const _Float16* __restrict__ q, const _Float16* __restrict__ kv,
    _Float16* __restrict__ cat) {
  __shared__ float    Sl[16][KPAD];
  __shared__ _Float16 Sp[16][KPAD];
  const int lane = threadIdx.x;
  const int qt = blockIdx.x, h = blockIdx.y, b = blockIdx.z;
  const float scale = 0.125f;                       // 64^-0.5

  const _Float16* qp = q + ((size_t)(b * NN + qt * 16)) * CC + h * HD;
  const v16h a0 = fragA_ld(qp, CC);
  const v16h a1 = fragA_ld(qp + 32, CC);

  const _Float16* kbase = kv + (size_t)(b * NKEY) * (2 * CC) + h * HD;
  for (int mt = 0; mt < KPAD / 16; ++mt) {
    const int m0 = mt * 16;
    const v16h b0 = fragB_nk_guard(kbase + (size_t)m0 * (2 * CC),      2 * CC, m0, NKEY);
    const v16h b1 = fragB_nk_guard(kbase + (size_t)m0 * (2 * CC) + 32, 2 * CC, m0, NKEY);
    v8f acc = zero8();
    acc = wmma_f16(a0, b0, acc);
    acc = wmma_f16(a1, b1, acc);
    const int nk = m0 + (lane & 15);
#pragma unroll
    for (int r = 0; r < 8; ++r) {
      const int mq = r + ((lane >> 4) << 3);
      Sl[mq][nk] = (nk < NKEY) ? acc[r] * scale : -1e30f;
    }
  }
  __syncthreads();

  if (lane < 16) {
    float mx = -1e30f;
    for (int m = 0; m < KPAD; ++m) mx = fmaxf(mx, Sl[lane][m]);
    float sum = 0.f;
    for (int m = 0; m < KPAD; ++m) { const float e = __expf(Sl[lane][m] - mx); Sl[lane][m] = e; sum += e; }
    const float inv = 1.f / sum;
    for (int m = 0; m < KPAD; ++m) Sp[lane][m] = (_Float16)(Sl[lane][m] * inv);
  }
  __syncthreads();

  const _Float16* vbase = kv + (size_t)(b * NKEY) * (2 * CC) + CC + h * HD;
  for (int nt = 0; nt < HD / 16; ++nt) {
    const int d0 = nt * 16;
    v8f acc = zero8();
    for (int kk = 0; kk < KPAD; kk += 32) {
      v16h a = fragA_ld(&Sp[0][0] + kk, KPAD);
      v16h v = fragB_kn_guard(vbase + (size_t)kk * (2 * CC) + d0, 2 * CC, kk, NKEY);
      acc = wmma_f16(a, v, acc);
    }
    const int n = lane & 15;
#pragma unroll
    for (int r = 0; r < 8; ++r) {
      const int mq = r + ((lane >> 4) << 3);
      const size_t row = (size_t)b * NN + qt * 16 + mq;
      cat[row * (CC + C4) + h * HD + d0 + n] = (_Float16)acc[r];
    }
  }
}

// ---------------------------------------------------------------------------
// Elementwise / reshape kernels
// ---------------------------------------------------------------------------
__global__ void cast_f32_f16(const float* __restrict__ s, _Float16* __restrict__ d, size_t n) {
  const size_t i = (size_t)blockIdx.x * 256 + threadIdx.x;
  if (i < n) d[i] = (_Float16)s[i];
}

// (O=512, I=512, taps) f32 -> (taps, O, I) f16
__global__ void repack_w(const float* __restrict__ src, _Float16* __restrict__ dst,
                         int taps, size_t total) {
  const size_t i = (size_t)blockIdx.x * 256 + threadIdx.x;
  if (i >= total) return;
  const int ic = (int)(i % CC);
  const int o  = (int)((i / CC) % CC);
  const int t  = (int)(i / ((size_t)CC * CC));
  dst[i] = (_Float16)src[((size_t)o * CC + ic) * taps + t];
}

// Haar DWT: r (B,56,56,128) f16 NHWC -> (B,28,28,512) f16 NHWC [ll,hl,lh,hh]
__global__ void dwt_haar(const _Float16* __restrict__ r, _Float16* __restrict__ out) {
  const size_t idx = (size_t)blockIdx.x * 256 + threadIdx.x;
  if (idx >= (size_t)BATCH * H2 * H2 * C4) return;
  const int c = (int)(idx & (C4 - 1));
  const int x = (int)((idx >> 7) % H2);
  const int y = (int)(((idx >> 7) / H2) % H2);
  const int b = (int)(idx / ((size_t)H2 * H2 * C4));
  const _Float16* base = r + (size_t)b * HH * WW * C4;
  const float x1 = (float)base[((size_t)(2 * y)     * WW + 2 * x)     * C4 + c] * 0.5f;
  const float x2 = (float)base[((size_t)(2 * y + 1) * WW + 2 * x)     * C4 + c] * 0.5f;
  const float x3 = (float)base[((size_t)(2 * y)     * WW + 2 * x + 1) * C4 + c] * 0.5f;
  const float x4 = (float)base[((size_t)(2 * y + 1) * WW + 2 * x + 1) * C4 + c] * 0.5f;
  _Float16* o = out + (((size_t)b * H2 + y) * H2 + x) * CC + c;
  o[0 * C4] = (_Float16)( x1 + x2 + x3 + x4);
  o[1 * C4] = (_Float16)(-x1 - x2 + x3 + x4);
  o[2 * C4] = (_Float16)(-x1 + x2 - x3 + x4);
  o[3 * C4] = (_Float16)( x1 - x2 - x3 + x4);
}

// Haar IDWT: (B,28,28,512) f16 -> concat buffer columns [512,640) (f16)
__global__ void idwt_haar(const _Float16* __restrict__ d, _Float16* __restrict__ cat) {
  const size_t idx = (size_t)blockIdx.x * 256 + threadIdx.x;
  if (idx >= (size_t)BATCH * H2 * H2 * C4) return;
  const int c = (int)(idx & (C4 - 1));
  const int x = (int)((idx >> 7) % H2);
  const int y = (int)(((idx >> 7) / H2) % H2);
  const int b = (int)(idx / ((size_t)H2 * H2 * C4));
  const _Float16* i = d + (((size_t)b * H2 + y) * H2 + x) * CC + c;
  const float ll = (float)i[0 * C4], hl = (float)i[1 * C4];
  const float lh = (float)i[2 * C4], hh = (float)i[3 * C4];
  const float p1 = (ll - hl - lh + hh) * 0.5f;
  const float p2 = (ll - hl + lh - hh) * 0.5f;
  const float p3 = (ll + hl - lh - hh) * 0.5f;
  const float p4 = (ll + hl + lh + hh) * 0.5f;
  const size_t ldcat = CC + C4;
#define WRI(yy, xx, v)                                                    \
  cat[((size_t)b * NN + (size_t)(yy) * WW + (xx)) * ldcat + CC + c] = (_Float16)(v)
  WRI(2 * y,     2 * x,     p1);
  WRI(2 * y,     2 * x + 1, p3);
  WRI(2 * y + 1, 2 * x,     p2);
  WRI(2 * y + 1, 2 * x + 1, p4);
#undef WRI
}

// LayerNorm over 512 channels, one block (128 thr) per row -> f16
__global__ __launch_bounds__(128) void layernorm_rows(
    const float* __restrict__ in, const float* __restrict__ g,
    const float* __restrict__ b, _Float16* __restrict__ out) {
  __shared__ float ss[128], sq[128];
  const int row = blockIdx.x, t = threadIdx.x;
  const float* p = in + (size_t)row * CC;
  float v[4], s = 0.f, q2 = 0.f;
#pragma unroll
  for (int j = 0; j < 4; ++j) { v[j] = p[t + 128 * j]; s += v[j]; q2 += v[j] * v[j]; }
  ss[t] = s; sq[t] = q2;
  __syncthreads();
  for (int st = 64; st > 0; st >>= 1) {
    if (t < st) { ss[t] += ss[t + st]; sq[t] += sq[t + st]; }
    __syncthreads();
  }
  const float mu   = ss[0] * (1.f / CC);
  const float var  = sq[0] * (1.f / CC) - mu * mu;
  const float rinv = rsqrtf(var + EPS_LN);
  _Float16* o = out + (size_t)row * CC;
#pragma unroll
  for (int j = 0; j < 4; ++j) {
    const int c = t + 128 * j;
    o[c] = (_Float16)((v[j] - mu) * rinv * g[c] + b[c]);
  }
}

// ---------------------------------------------------------------------------
extern "C" void kernel_launch(void* const* d_in, const int* in_sizes, int n_in,
                              void* d_out, int out_size, void* d_ws, size_t ws_size,
                              hipStream_t stream) {
  (void)in_sizes; (void)n_in; (void)out_size; (void)ws_size;
  const float* x          = (const float*)d_in[0];
  const float* reduce_w   = (const float*)d_in[3];
  const float* reduce_b   = (const float*)d_in[4];
  const float* reduce_g   = (const float*)d_in[5];
  const float* reduce_be  = (const float*)d_in[6];
  const float* filter_w   = (const float*)d_in[7];
  const float* filter_b   = (const float*)d_in[8];
  const float* filter_g   = (const float*)d_in[9];
  const float* filter_be  = (const float*)d_in[10];
  const float* q_w        = (const float*)d_in[11];
  const float* q_b        = (const float*)d_in[12];
  const float* ln_g       = (const float*)d_in[13];
  const float* ln_b       = (const float*)d_in[14];
  const float* kv_w       = (const float*)d_in[15];
  const float* kv_b       = (const float*)d_in[16];
  const float* kve_w      = (const float*)d_in[17];
  const float* kve_b      = (const float*)d_in[18];
  const float* proj_w     = (const float*)d_in[19];
  const float* proj_b     = (const float*)d_in[20];
  float* out = (float*)d_out;

  const size_t NQ = (size_t)BATCH * NN;       // 25088 rows
  char* ws = (char*)d_ws;

  // Region 0: x16 (25,690,112) + r16 (6,422,528) -- later reused as cat
  // (exactly 25088*640*2 = 32,112,640 bytes; x16/r16 are dead by then).
  _Float16* x16  = (_Float16*)(ws + 0);
  _Float16* r16  = (_Float16*)(ws + NQ * CC * 2);
  _Float16* cat  = (_Float16*)(ws + 0);
  size_t off = NQ * (CC + C4) * 2;
  auto alloc = [&](size_t bytes) { void* p = ws + off; off += (bytes + 255) & ~(size_t)255; return p; };
  _Float16* qbuf   = (_Float16*)alloc(NQ * CC * 2);
  _Float16* kvbuf  = (_Float16*)alloc((size_t)BATCH * NKEY * 2 * CC * 2);
  _Float16* dwtin  = (_Float16*)alloc((size_t)BATCH * H2 * H2 * CC * 2);
  _Float16* xdwt16 = (_Float16*)alloc((size_t)BATCH * H2 * H2 * CC * 2);
  float*    kvin   = (float*)   alloc((size_t)BATCH * NKEY * CC * 4);
  _Float16* kvn    = (_Float16*)alloc((size_t)BATCH * NKEY * CC * 2);
  _Float16* qw16   = (_Float16*)alloc((size_t)CC * CC * 2);
  _Float16* kvw16  = (_Float16*)alloc((size_t)2 * CC * CC * 2);
  _Float16* pjw16  = (_Float16*)alloc((size_t)CC * (CC + C4) * 2);
  _Float16* rdw16  = (_Float16*)alloc((size_t)C4 * CC * 2);
  _Float16* fw16   = (_Float16*)alloc((size_t)9 * CC * CC * 2);
  _Float16* kew16  = (_Float16*)alloc((size_t)4 * CC * CC * 2);

  auto blks = [](size_t n) { return (unsigned)((n + 255) / 256); };

  // -- weight & activation casts / repacks
  cast_f32_f16<<<blks(NQ * CC), 256, 0, stream>>>(x, x16, NQ * CC);
  cast_f32_f16<<<blks((size_t)CC * CC), 256, 0, stream>>>(q_w, qw16, (size_t)CC * CC);
  cast_f32_f16<<<blks((size_t)2 * CC * CC), 256, 0, stream>>>(kv_w, kvw16, (size_t)2 * CC * CC);
  cast_f32_f16<<<blks((size_t)CC * (CC + C4)), 256, 0, stream>>>(proj_w, pjw16, (size_t)CC * (CC + C4));
  cast_f32_f16<<<blks((size_t)C4 * CC), 256, 0, stream>>>(reduce_w, rdw16, (size_t)C4 * CC);
  repack_w<<<blks((size_t)9 * CC * CC), 256, 0, stream>>>(filter_w, fw16, 9, (size_t)9 * CC * CC);
  repack_w<<<blks((size_t)4 * CC * CC), 256, 0, stream>>>(kve_w, kew16, 4, (size_t)4 * CC * CC);

  // -- q = x @ q_w.T + q_b  (f16 out)
  gemm_xwt<<<dim3(NQ / 16, CC / 64), 128, 0, stream>>>(
      x16, qw16, q_b, nullptr, nullptr, nullptr, qbuf, CC, CC, 1);

  // -- r = relu(bn(conv1x1(x))) as GEMM (f16 out, NHWC)
  gemm_xwt<<<dim3(NQ / 16, C4 / 64), 128, 0, stream>>>(
      x16, rdw16, reduce_b, reduce_g, reduce_be, nullptr, r16, CC, C4, 2);

  // -- DWT -> (B,28,28,512)
  dwt_haar<<<blks((size_t)BATCH * H2 * H2 * C4), 256, 0, stream>>>(r16, dwtin);

  // -- 3x3 conv + BN + ReLU -> x_dwt16
  conv3x3_bn_relu<<<dim3((H2 * H2) / 16, CC / 64, BATCH), 128, 0, stream>>>(
      dwtin, fw16, filter_b, filter_g, filter_be, xdwt16);

  // -- IDWT -> concat columns [512,640)   (x16/r16 are dead from here on)
  idwt_haar<<<blks((size_t)BATCH * H2 * H2 * C4), 256, 0, stream>>>(xdwt16, cat);

  // -- 2x2/s2 conv -> kv_in (f32)
  conv2x2_s2_bias<<<dim3((NKEY + 15) / 16, CC / 64, BATCH), 128, 0, stream>>>(
      xdwt16, kew16, kve_b, kvin);

  // -- LayerNorm -> kv_n (f16)
  layernorm_rows<<<BATCH * NKEY, 128, 0, stream>>>(kvin, ln_g, ln_b, kvn);

  // -- kv = kv_n @ kv_w.T + kv_b  (f16, 1024 cols: [K | V])
  gemm_xwt<<<dim3((BATCH * NKEY) / 16, (2 * CC) / 64), 128, 0, stream>>>(
      kvn, kvw16, kv_b, nullptr, nullptr, nullptr, kvbuf, CC, 2 * CC, 1);

  // -- fused attention -> concat columns [0,512)
  attention_fused<<<dim3(NN / 16, NHEAD, BATCH), 32, 0, stream>>>(qbuf, kvbuf, cat);

  // -- out = concat @ proj_w.T + proj_b  (f32)
  gemm_xwt<<<dim3(NQ / 16, CC / 64), 128, 0, stream>>>(
      cat, pjw16, proj_b, nullptr, nullptr, out, nullptr, CC + C4, CC, 0);
}
